// Decoder_81930796139068
// MI455X (gfx1250) — matrix operand order repaired and checked by reading, not compiled, not run on hardware
//
#include <hip/hip_runtime.h>
#include <hip/hip_bf16.h>

// ---------------------------------------------------------------------------
// Types for CDNA5 WMMA (wave32, 16x16x32 bf16 -> f32)
// ---------------------------------------------------------------------------
typedef __bf16 bf16;
typedef __attribute__((ext_vector_type(16))) __bf16 v16bf;
typedef __attribute__((ext_vector_type(8)))  float  v8f;

#define V_  32000
#define E_  256
#define H_  512
#define S_  128
#define T_  64
#define B_  64

__device__ __forceinline__ bf16 to_bf16(float f) {
  unsigned int u = __builtin_bit_cast(unsigned int, f);
  unsigned int r = (u + 0x7FFFu + ((u >> 16) & 1u)) >> 16;   // RNE
  unsigned short s = (unsigned short)r;
  return __builtin_bit_cast(bf16, s);
}

__device__ __forceinline__ float sigmoidf_(float x) { return 1.f / (1.f + __expf(-x)); }

// Compiler scheduling fence: keeps the software pipeline's issue order.
#define SCHED_FENCE() asm volatile("" ::: "memory")

union Frag { v16bf v; uint4 q[2]; };

// A-matrix 16x32 bf16 fragment (ISA 7.12.2): lane m=lane&15; lanes 0-15 hold
// K = k0+0..7 (elems 0..7) and k0+16..23 (elems 8..15); lanes 16-31 hold +8.
__device__ __forceinline__ v16bf load_a_frag(const bf16* base, int ld, int row0,
                                             int k0, int lane) {
  int m  = lane & 15;
  int h8 = (lane >> 4) << 3;
  const bf16* p = base + (size_t)(row0 + m) * ld + k0 + h8;
  Frag f;
  f.q[0] = *(const uint4*)p;          // elems 0..7
  f.q[1] = *(const uint4*)(p + 16);   // elems 8..15 (K + 16)
  return f.v;
}

// B-matrix 32x16 bf16 fragment: lane n=lane&15 holds K = k0+8*(lane>>4)+0..7
// (elems 0..7) and +16 (elems 8..15) of column n0+n. Weights stored N-major [N][K].
__device__ __forceinline__ v16bf load_b_frag(const bf16* wt, int ld, int n0,
                                             int k0, int lane) {
  int n   = lane & 15;
  int k16 = (lane >> 4) << 3;
  const bf16* p = wt + (size_t)(n0 + n) * ld + k0 + k16;
  Frag f;
  f.q[0] = *(const uint4*)p;
  f.q[1] = *(const uint4*)(p + 16);
  return f.v;
}

__device__ __forceinline__ v8f wmma_bf16(v16bf a, v16bf b, v8f c) {
  return __builtin_amdgcn_wmma_f32_16x16x32_bf16(false, a, false, b,
                                                 (short)0, c, false, false);
}

// ---------------------------------------------------------------------------
// Common GEMM mainloop: 1 M-tile x 4 N-tiles per wave, rotating 3-deep
// register pipeline. Loads for chunk c+2 are issued (and fenced) before the
// WMMAs of chunk c, into the slot freed at chunk c-1, so every WMMA waits on
// loads that are two chunks old.
// ---------------------------------------------------------------------------
template <int K, bool PF>
__device__ __forceinline__ void gemm4_loop(const bf16* __restrict__ A, int lda,
                                           int row0,
                                           const bf16* __restrict__ Bt, int ldb,
                                           const int nbase[4], int lane,
                                           v8f acc[4]) {
  constexpr int NC = K / 32;
  v16bf aB[3];
  v16bf bB[3][4];
#pragma unroll
  for (int c = 0; c < 2; c++) {
    aB[c] = load_a_frag(A, lda, row0, c * 32, lane);
#pragma unroll
    for (int j = 0; j < 4; j++)
      bB[c][j] = load_b_frag(Bt, ldb, nbase[j], c * 32, lane);
  }
  SCHED_FENCE();                       // pin prologue loads before the loop
#pragma unroll
  for (int c = 0; c < NC; c++) {
    if (c + 2 < NC) {
      const int kn   = (c + 2) * 32;
      const int slot = (c + 2) % 3;
      aB[slot] = load_a_frag(A, lda, row0, kn, lane);
#pragma unroll
      for (int j = 0; j < 4; j++)
        bB[slot][j] = load_b_frag(Bt, ldb, nbase[j], kn, lane);
      if (PF && (c & 1) == 0 && kn + 224 < K) {   // gfx1250 global_prefetch_b8
#pragma unroll
        for (int j = 0; j < 4; j++)
          __builtin_prefetch(Bt + (size_t)(nbase[j] + (lane & 15)) * ldb + kn + 224, 0, 1);
      }
      SCHED_FENCE();                   // loads for c+2 issued before WMMAs of c
    }
    const int cur = c % 3;
#pragma unroll
    for (int j = 0; j < 4; j++) acc[j] = wmma_bf16(aB[cur], bB[cur][j], acc[j]);
  }
}

// ---------------------------------------------------------------------------
// Prep kernels: weight transpose/convert, state init
// ---------------------------------------------------------------------------

// WT[n*K + k] = (k < Kih ? Wih[k][n] : Whh[k-Kih][n]) as bf16.
__global__ void k_prep_wcat(const float* __restrict__ Wih, int Kih,
                            const float* __restrict__ Whh, int Khh,
                            int N, bf16* __restrict__ WT) {
  int K = Kih + Khh;
  size_t total  = (size_t)N * K;
  size_t stride = (size_t)gridDim.x * blockDim.x;
  for (size_t idx = (size_t)blockIdx.x * blockDim.x + threadIdx.x; idx < total;
       idx += stride) {
    int n = (int)(idx / K);
    int k = (int)(idx % K);
    float v = (k < Kih) ? Wih[(size_t)k * N + n] : Whh[(size_t)(k - Kih) * N + n];
    WT[idx] = to_bf16(v);
  }
}

// mem_bf16[b][s][k] = memory_bank[s][b][k]
__global__ void k_prep_mem(const float* __restrict__ mb, bf16* __restrict__ out) {
  size_t total  = (size_t)B_ * S_ * H_;
  size_t stride = (size_t)gridDim.x * blockDim.x;
  for (size_t idx = (size_t)blockIdx.x * blockDim.x + threadIdx.x; idx < total;
       idx += stride) {
    int b = (int)(idx >> 16);          // / (S_*H_)
    int s = (int)((idx >> 9) & (S_ - 1));
    int k = (int)(idx & (H_ - 1));
    out[idx] = to_bf16(mb[((size_t)s * B_ + b) * H_ + k]);
  }
}

__global__ void k_prep_bias(const float* __restrict__ bih,
                            const float* __restrict__ bhh,
                            float* __restrict__ bsum) {
  int idx = blockIdx.x * blockDim.x + threadIdx.x;
  if (idx < 4 * H_) bsum[idx] = bih[idx] + bhh[idx];
}

// fix(): layer l hidden = concat(enc[2l], enc[2l+1]) along H
__global__ void k_init_state(const float* __restrict__ enc_h,
                             const float* __restrict__ enc_c,
                             float* __restrict__ hA, float* __restrict__ hB,
                             float* __restrict__ cA, float* __restrict__ cB) {
  int idx = blockIdx.x * blockDim.x + threadIdx.x;
  if (idx >= B_ * H_) return;
  int b = idx / H_, h = idx % H_;
  int dir = (h < (H_ / 2)) ? 0 : 1;
  int hh  = h & (H_ / 2 - 1);
  hA[idx] = enc_h[((size_t)(0 + dir) * B_ + b) * (H_ / 2) + hh];
  hB[idx] = enc_h[((size_t)(2 + dir) * B_ + b) * (H_ / 2) + hh];
  cA[idx] = enc_c[((size_t)(0 + dir) * B_ + b) * (H_ / 2) + hh];
  cB[idx] = enc_c[((size_t)(2 + dir) * B_ + b) * (H_ / 2) + hh];
}

// ---------------------------------------------------------------------------
// memW[b*S+s][0:1024] = mem[b,s,:] @ W1[0:H, :]   (WMMA, time-invariant)
// grid (4, 512), block 128 (4 waves); each wave does 4 N-tiles of one M-tile.
// ---------------------------------------------------------------------------
__global__ void __launch_bounds__(128, 1)
k_memproj(const bf16* __restrict__ membf, const bf16* __restrict__ w1topT,
          float* __restrict__ memW) {
  int lane  = threadIdx.x & 31;
  int wid   = threadIdx.x >> 5;
  int wtile = blockIdx.x * 4 + wid;   // 0..15
  int row0  = blockIdx.y * 16;        // 0..8191
  int nbase[4];
#pragma unroll
  for (int j = 0; j < 4; j++) nbase[j] = (wtile * 4 + j) * 16;

  v8f acc[4];
#pragma unroll
  for (int j = 0; j < 4; j++) acc[j] = (v8f){0.f,0.f,0.f,0.f,0.f,0.f,0.f,0.f};

  gemm4_loop<H_, true>(membf, H_, row0, w1topT, H_, nbase, lane, acc);

  int m0 = (lane >> 4) << 3, n = lane & 15;
#pragma unroll
  for (int j = 0; j < 4; j++) {
#pragma unroll
    for (int r = 0; r < 8; r++)
      memW[(size_t)(row0 + m0 + r) * (2 * H_) + nbase[j] + n] = acc[j][r];
  }
}

// ---------------------------------------------------------------------------
// Per-step attention (one block per batch element b)
// ---------------------------------------------------------------------------
__global__ void __launch_bounds__(256)
k_attn(const float* __restrict__ hA, const float* __restrict__ hB,
       const float* __restrict__ memW, const float* __restrict__ memory_bank,
       const float* __restrict__ W1, const float* __restrict__ w2,
       const int* __restrict__ valid_lens, const float* __restrict__ embedding,
       const int* __restrict__ tgt, int t,
       bf16* __restrict__ xcat, bf16* __restrict__ xh1,
       float* __restrict__ attns) {
  __shared__ float q[H_];
  __shared__ float qp[2 * H_];
  __shared__ float sc[S_];
  int b = blockIdx.x;
  int tid = threadIdx.x;
  int lane = tid & 31, wid = tid >> 5;

  for (int i = tid; i < H_; i += 256) q[i] = hB[(size_t)b * H_ + i];
  __syncthreads();

  // qproj = q @ W1[H:2H, :]
  for (int j = tid; j < 2 * H_; j += 256) {
    float acc = 0.f;
    for (int k = 0; k < H_; k++)
      acc += q[k] * W1[(size_t)(H_ + k) * (2 * H_) + j];
    qp[j] = acc;
  }
  __syncthreads();

  // scores[s] = sum_j w2[j] * tanh(memW[b,s,j] + qp[j])
  const float* mrowbase = memW + (size_t)b * S_ * (2 * H_);
  for (int s = wid; s < S_; s += 8) {
    const float* mrow = mrowbase + (size_t)s * (2 * H_);
    float part = 0.f;
    for (int j = lane; j < 2 * H_; j += 32)
      part += w2[j] * tanhf(mrow[j] + qp[j]);
    for (int off = 16; off > 0; off >>= 1) part += __shfl_down(part, off, 32);
    if (lane == 0) sc[s] = part;
  }
  __syncthreads();

  int vl = valid_lens[b];
  if (tid < S_ && tid >= vl) sc[tid] = -1e6f;
  __syncthreads();

  if (wid == 0) {                 // masked softmax over S=128 by wave 0
    float m = -3.4e38f;
    for (int s = lane; s < S_; s += 32) m = fmaxf(m, sc[s]);
    for (int off = 16; off > 0; off >>= 1) m = fmaxf(m, __shfl_down(m, off, 32));
    m = __shfl(m, 0, 32);
    float sum = 0.f;
    for (int s = lane; s < S_; s += 32) { float e = __expf(sc[s] - m); sc[s] = e; sum += e; }
    for (int off = 16; off > 0; off >>= 1) sum += __shfl_down(sum, off, 32);
    sum = __shfl(sum, 0, 32);
    float inv = 1.f / sum;
    for (int s = lane; s < S_; s += 32) sc[s] *= inv;
  }
  __syncthreads();

  for (int s = tid; s < S_; s += 256)
    attns[(size_t)t * B_ * S_ + (size_t)b * S_ + s] = sc[s];

  int tok = (t == 0) ? 1 : tgt[(t - 1) * B_ + b];
  if (tid < E_)
    xcat[(size_t)b * (E_ + 2 * H_) + tid] =
        to_bf16(embedding[(size_t)tok * E_ + tid]);

  for (int h = tid; h < H_; h += 256) {
    float acc = 0.f;
    for (int s = 0; s < S_; s++)
      acc += sc[s] * memory_bank[((size_t)s * B_ + b) * H_ + h];
    xcat[(size_t)b * (E_ + 2 * H_) + E_ + h]      = to_bf16(acc);
    xcat[(size_t)b * (E_ + 2 * H_) + E_ + H_ + h] = to_bf16(hA[(size_t)b * H_ + h]);
    xh1 [(size_t)b * (2 * H_) + H_ + h]           = to_bf16(hB[(size_t)b * H_ + h]);
  }
}

// ---------------------------------------------------------------------------
// Fused LSTM cell: gates = xin[64,K] @ WT[2048,K]^T (+bsum), then cell update.
// grid (32, 4), block 32 (one wave -> huge VGPR budget for the pipeline).
// ---------------------------------------------------------------------------
template <int K>
__global__ void __launch_bounds__(32, 1)
k_lstm(const bf16* __restrict__ xin,
       const bf16* __restrict__ WT, const float* __restrict__ bsum,
       float* __restrict__ hstate, float* __restrict__ cstate,
       bf16* __restrict__ hbf, int hbf_ld) {
  int lane  = threadIdx.x;
  int hcol0 = blockIdx.x * 16;
  int row0  = blockIdx.y * 16;
  int nbase[4];
#pragma unroll
  for (int g = 0; g < 4; g++) nbase[g] = g * H_ + hcol0;

  v8f acc[4];
#pragma unroll
  for (int g = 0; g < 4; g++) acc[g] = (v8f){0.f,0.f,0.f,0.f,0.f,0.f,0.f,0.f};

  gemm4_loop<K, false>(xin, K, row0, WT, K, nbase, lane, acc);

  int m0 = (lane >> 4) << 3, n = lane & 15;
  int hcol = hcol0 + n;
  float bi = bsum[0 * H_ + hcol], bf_ = bsum[1 * H_ + hcol];
  float bg = bsum[2 * H_ + hcol], bo  = bsum[3 * H_ + hcol];
#pragma unroll
  for (int r = 0; r < 8; r++) {
    int bb = row0 + m0 + r;
    float iv = sigmoidf_(acc[0][r] + bi);
    float fv = sigmoidf_(acc[1][r] + bf_);
    float gv = tanhf   (acc[2][r] + bg);
    float ov = sigmoidf_(acc[3][r] + bo);
    size_t sidx = (size_t)bb * H_ + hcol;
    float cn = fv * cstate[sidx] + iv * gv;
    float hn = ov * tanhf(cn);
    cstate[sidx] = cn;
    hstate[sidx] = hn;
    hbf[(size_t)bb * hbf_ld + hcol] = to_bf16(hn);
  }
}

// ---------------------------------------------------------------------------
// dec_outs = outs[4096,512] @ Wd[512,32000] + bd  (WMMA, store-bound)
// grid (125, 256), block 128 (4 waves); wave does 4 N-tiles of one M-tile.
// ---------------------------------------------------------------------------
__global__ void __launch_bounds__(128, 1)
k_out(const bf16* __restrict__ outsbf, const bf16* __restrict__ wdT,
      const float* __restrict__ bd, float* __restrict__ dec) {
  int lane  = threadIdx.x & 31;
  int wid   = threadIdx.x >> 5;
  int wtile = blockIdx.x * 4 + wid;   // 0..499
  int row0  = blockIdx.y * 16;        // 0..4095
  int nbase[4];
#pragma unroll
  for (int j = 0; j < 4; j++) nbase[j] = (wtile * 4 + j) * 16;

  v8f acc[4];
#pragma unroll
  for (int j = 0; j < 4; j++) acc[j] = (v8f){0.f,0.f,0.f,0.f,0.f,0.f,0.f,0.f};

  gemm4_loop<H_, true>(outsbf, H_, row0, wdT, H_, nbase, lane, acc);

  int m0 = (lane >> 4) << 3, n = lane & 15;
#pragma unroll
  for (int j = 0; j < 4; j++) {
    float bias = bd[nbase[j] + n];
#pragma unroll
    for (int r = 0; r < 8; r++)
      dec[(size_t)(row0 + m0 + r) * V_ + nbase[j] + n] = acc[j][r] + bias;
  }
}

__global__ void k_copy_state(const float* __restrict__ hA, const float* __restrict__ hB,
                             const float* __restrict__ cA, const float* __restrict__ cB,
                             float* __restrict__ dst) {
  int idx = blockIdx.x * blockDim.x + threadIdx.x;
  if (idx >= B_ * H_) return;
  dst[idx]                 = hA[idx];
  dst[B_ * H_ + idx]       = hB[idx];
  dst[2 * B_ * H_ + idx]   = cA[idx];
  dst[3 * B_ * H_ + idx]   = cB[idx];
}

// ---------------------------------------------------------------------------
// Host launcher
// ---------------------------------------------------------------------------
static inline size_t alignup(size_t x) { return (x + 255) & ~(size_t)255; }

extern "C" void kernel_launch(void* const* d_in, const int* in_sizes, int n_in,
                              void* d_out, int out_size, void* d_ws, size_t ws_size,
                              hipStream_t stream) {
  (void)in_sizes; (void)n_in; (void)out_size; (void)ws_size;
  const int*   tgt         = (const int*)  d_in[0];
  const float* memory_bank = (const float*)d_in[1];
  const int*   valid_lens  = (const int*)  d_in[2];
  const float* enc_h       = (const float*)d_in[3];
  const float* enc_c       = (const float*)d_in[4];
  const float* embedding   = (const float*)d_in[5];
  const float* W1          = (const float*)d_in[6];
  const float* w2          = (const float*)d_in[7];
  const float* Wih0        = (const float*)d_in[8];
  const float* Whh0        = (const float*)d_in[9];
  const float* bih0        = (const float*)d_in[10];
  const float* bhh0        = (const float*)d_in[11];
  const float* Wih1        = (const float*)d_in[12];
  const float* Whh1        = (const float*)d_in[13];
  const float* bih1        = (const float*)d_in[14];
  const float* bhh1        = (const float*)d_in[15];
  const float* Wd          = (const float*)d_in[16];
  const float* bd          = (const float*)d_in[17];
  float* out = (float*)d_out;
  char*  ws  = (char*)d_ws;

  // output layout: dec_outs [T,B,V], hT [2,B,H], cT [2,B,H], attns [T,B,S]
  const size_t O_DEC  = 0;
  const size_t O_HT   = (size_t)T_ * B_ * V_;
  const size_t O_CT   = O_HT + 2 * (size_t)B_ * H_;
  const size_t O_ATTN = O_CT + 2 * (size_t)B_ * H_;

  // workspace layout
  size_t off = 0;
  bf16*  wcat0T = (bf16*)(ws + off); off = alignup(off + (size_t)4 * H_ * (E_ + 2 * H_) * 2); // [2048][1280]
  bf16*  wcat1T = (bf16*)(ws + off); off = alignup(off + (size_t)4 * H_ * (2 * H_) * 2);      // [2048][1024]
  bf16*  wdT    = (bf16*)(ws + off); off = alignup(off + (size_t)V_ * H_ * 2);                // [32000][512]
  bf16*  w1topT = (bf16*)(ws + off); off = alignup(off + (size_t)(2 * H_) * H_ * 2);          // [1024][512]
  bf16*  membf  = (bf16*)(ws + off); off = alignup(off + (size_t)B_ * S_ * H_ * 2);           // [8192][512]
  float* memW   = (float*)(ws + off); off = alignup(off + (size_t)B_ * S_ * (2 * H_) * 4);    // [8192][1024]
  float* bsum0  = (float*)(ws + off); off = alignup(off + 4 * H_ * 4);
  float* bsum1  = (float*)(ws + off); off = alignup(off + 4 * H_ * 4);
  bf16*  xcat   = (bf16*)(ws + off); off = alignup(off + (size_t)B_ * (E_ + 2 * H_) * 2);     // [64][1280]
  bf16*  xh1    = (bf16*)(ws + off); off = alignup(off + (size_t)B_ * (2 * H_) * 2);          // [64][1024]
  bf16*  outsbf = (bf16*)(ws + off); off = alignup(off + (size_t)T_ * B_ * H_ * 2);           // [4096][512]
  float* hA = (float*)(ws + off); off = alignup(off + (size_t)B_ * H_ * 4);
  float* hB = (float*)(ws + off); off = alignup(off + (size_t)B_ * H_ * 4);
  float* cA = (float*)(ws + off); off = alignup(off + (size_t)B_ * H_ * 4);
  float* cB = (float*)(ws + off); off = alignup(off + (size_t)B_ * H_ * 4);

  dim3 blk256(256);
  k_prep_wcat<<<dim3(2048), blk256, 0, stream>>>(Wih0, E_ + H_, Whh0, H_, 4 * H_, wcat0T);
  k_prep_wcat<<<dim3(2048), blk256, 0, stream>>>(Wih1, H_, Whh1, H_, 4 * H_, wcat1T);
  k_prep_wcat<<<dim3(8192), blk256, 0, stream>>>(Wd, H_, nullptr, 0, V_, wdT);
  k_prep_wcat<<<dim3(1024), blk256, 0, stream>>>(W1, H_, nullptr, 0, 2 * H_, w1topT);
  k_prep_mem <<<dim3(4096), blk256, 0, stream>>>(memory_bank, membf);
  k_prep_bias<<<dim3(8),    blk256, 0, stream>>>(bih0, bhh0, bsum0);
  k_prep_bias<<<dim3(8),    blk256, 0, stream>>>(bih1, bhh1, bsum1);
  k_init_state<<<dim3(128), blk256, 0, stream>>>(enc_h, enc_c, hA, hB, cA, cB);
  k_memproj<<<dim3(4, 512), dim3(128), 0, stream>>>(membf, w1topT, memW);

  for (int t = 0; t < T_; t++) {
    k_attn<<<dim3(B_), dim3(256), 0, stream>>>(hA, hB, memW, memory_bank, W1, w2,
                                               valid_lens, embedding, tgt, t,
                                               xcat, xh1, out + O_ATTN);
    k_lstm<E_ + 2 * H_><<<dim3(32, 4), dim3(32), 0, stream>>>(
        xcat, wcat0T, bsum0, hA, cA, xh1, 2 * H_);
    k_lstm<2 * H_><<<dim3(32, 4), dim3(32), 0, stream>>>(
        xh1, wcat1T, bsum1, hB, cB, outsbf + (size_t)t * B_ * H_, H_);
  }

  k_out<<<dim3(125, 256), dim3(128), 0, stream>>>(outsbf, wdT, bd, out + O_DEC);
  k_copy_state<<<dim3(128), blk256, 0, stream>>>(hA, hB, cA, cB, out + O_HT);
}
